// CrossAttention_88768384073954
// MI455X (gfx1250) — compile-verified, matrix operand order checked by reading, not run on hardware
//
#include <hip/hip_runtime.h>

// ---------------------------------------------------------------------------
// Cross-attention for MI455X (gfx1250, wave32, WMMA f32_16x16x32_f16).
//   q = x@Wq+bq (scale*log2e folded in), k = ctx@Wk+bk, v = ctx@Wv+bv
//   flash attention per (b,h) with exp2-domain online softmax
//   out = attn@Wo+bo
// Weights are pre-transposed/converted to f16 once; GEMM + attention stage
// their shared tiles into LDS with gfx1250 async loads (ASYNCcnt path).
// ---------------------------------------------------------------------------

typedef __attribute__((ext_vector_type(16))) _Float16 v16h;
typedef __attribute__((ext_vector_type(8)))  _Float16 v8h;
typedef __attribute__((ext_vector_type(8)))  float    v8f;
typedef __attribute__((ext_vector_type(4)))  float    v4f;

#define DIM    1024
#define NHEADS 16
#define HDIM   64
#define NTOK   8192                       // B * N = B * M
#define LOG2E  1.4426950408889634f
#define ATT_SCALE 0.125f                  // HEAD_DIM^-0.5

__device__ __forceinline__ v8f wmma_f16(v16h a, v16h b, v8f c) {
  return __builtin_amdgcn_wmma_f32_16x16x32_f16(false, a, false, b,
                                                (short)0, c, false, false);
}

// gfx1250 async copy: 16 bytes global -> LDS, tracked by ASYNCcnt.
__device__ __forceinline__ void async_b128(const _Float16* lds_dst,
                                           const _Float16* gsrc) {
  const unsigned off = (unsigned)(uintptr_t)lds_dst;   // LDS aperture: low 32 bits
  asm volatile("global_load_async_to_lds_b128 %0, %1, off"
               :: "v"(off), "v"(gsrc) : "memory");
}
__device__ __forceinline__ void wait_async0() {
  asm volatile("s_wait_asynccnt 0x0" ::: "memory");
}

// 16x32 f16 fragment (A layout; symmetric B layout) from a row-major f16
// source (global or LDS). CDNA5 VGPR striping:
//   lane&15 = row, e in [0,8): K = 8*(lane>>4)+e ; e in [8,16): K = 16+8*(lane>>4)+e-8
__device__ __forceinline__ v16h frag_f16(const _Float16* __restrict__ src, int ld) {
  const int lane = threadIdx.x & 31;
  const int r    = lane & 15;
  const int kh   = (lane >> 4) << 3;
  const _Float16* p = src + r * ld + kh;
  v8h lo = *reinterpret_cast<const v8h*>(p);
  v8h hi = *reinterpret_cast<const v8h*>(p + 16);
  v16h f;
#pragma unroll
  for (int i = 0; i < 8; ++i) { f[i] = lo[i]; f[8 + i] = hi[i]; }
  return f;
}

// Same fragment from an f32 source (activations), converted to f16.
__device__ __forceinline__ v16h frag_f32(const float* __restrict__ src, int ld) {
  const int lane = threadIdx.x & 31;
  const int r    = lane & 15;
  const int kh   = (lane >> 4) << 3;
  const float* p = src + r * ld + kh;
  v4f a0 = *reinterpret_cast<const v4f*>(p);
  v4f a1 = *reinterpret_cast<const v4f*>(p + 4);
  v4f b0 = *reinterpret_cast<const v4f*>(p + 16);
  v4f b1 = *reinterpret_cast<const v4f*>(p + 20);
  v16h f;
#pragma unroll
  for (int i = 0; i < 4; ++i) {
    f[i]      = (_Float16)a0[i];
    f[4 + i]  = (_Float16)a1[i];
    f[8 + i]  = (_Float16)b0[i];
    f[12 + i] = (_Float16)b1[i];
  }
  return f;
}

// ---------------------------------------------------------------------------
// One-time weight prep: W f32 [K x N]  ->  WT f16 [N x K]  (LDS tile transpose)
// ---------------------------------------------------------------------------
__global__ __launch_bounds__(256) void w_transpose_kernel(
    const float* __restrict__ W, _Float16* __restrict__ WT) {
  __shared__ _Float16 t[32][33];
  const int n0 = blockIdx.x * 32, k0 = blockIdx.y * 32;
  const int tx = threadIdx.x & 31, ty = threadIdx.x >> 5;   // 32 x 8
#pragma unroll
  for (int r = 0; r < 4; ++r)
    t[ty + r * 8][tx] = (_Float16)W[(size_t)(k0 + ty + r * 8) * DIM + n0 + tx];
  __syncthreads();
#pragma unroll
  for (int r = 0; r < 4; ++r)
    WT[(size_t)(n0 + ty + r * 8) * DIM + k0 + tx] = t[tx][ty + r * 8];
}

// ---------------------------------------------------------------------------
// GEMM: out = (A[R x 1024] @ W + bias) * scale, W given as WT f16 [N x K].
// Workgroup: 8 waves, 256x64 tile; wave tile 32x64 (8 accumulators).
// WT tile (64 cols x 32 k, 4KB) is async-copied to a double-buffered LDS
// tile once per workgroup per k-step; B frags are ds_load_b128 from LDS.
// OUT_MODE 0: f16 [R x 1024]   (Q/K/attn-in)
// OUT_MODE 1: f16 V transposed: VT[(b*1024 + col)*1024 + m]
// OUT_MODE 2: f32 [R x 1024]   (final output)
// ---------------------------------------------------------------------------
template <bool A_F16, int OUT_MODE>
__global__ __launch_bounds__(256) void gemm1024_kernel(
    const void* __restrict__ Av, const _Float16* __restrict__ WT,
    const float* __restrict__ bias, void* __restrict__ outv, float scale) {
  __shared__ __align__(16) _Float16 wbuf[2][64][32];   // [buf][col][k]

  const int wave    = threadIdx.x >> 5;
  const int lane    = threadIdx.x & 31;
  const int rowbase = blockIdx.y * 256 + wave * 32;
  const int colbase = blockIdx.x * 64;

  // async-staging assignment: thread t copies 8 halfs of the WT tile
  const int scol = threadIdx.x >> 2;            // 0..63
  const int skk  = (threadIdx.x & 3) * 8;       // 0,8,16,24
  const _Float16* wsrc = WT + (size_t)(colbase + scol) * DIM + skk;

  async_b128(&wbuf[0][scol][skk], wsrc);        // prologue: k-step 0
  wait_async0();
  __syncthreads();

  v8f acc[2][4] = {};
  for (int k0 = 0; k0 < DIM; k0 += 32) {
    const int p = (k0 >> 5) & 1;
    if (k0 + 32 < DIM)                          // kick next tile into other buffer
      async_b128(&wbuf[p ^ 1][scol][skk], wsrc + k0 + 32);

    v16h a0, a1;
    if (A_F16) {
      a0 = frag_f16((const _Float16*)Av + (size_t)rowbase * DIM + k0, DIM);
      a1 = frag_f16((const _Float16*)Av + (size_t)(rowbase + 16) * DIM + k0, DIM);
    } else {
      a0 = frag_f32((const float*)Av + (size_t)rowbase * DIM + k0, DIM);
      a1 = frag_f32((const float*)Av + (size_t)(rowbase + 16) * DIM + k0, DIM);
    }
#pragma unroll
    for (int j = 0; j < 4; ++j) {
      v16h b = frag_f16(&wbuf[p][j * 16][0], 32);
      acc[0][j] = wmma_f16(a0, b, acc[0][j]);
      acc[1][j] = wmma_f16(a1, b, acc[1][j]);
    }
    wait_async0();
    __syncthreads();
  }

  const int hi = lane >> 4;
  const int c  = lane & 15;
#pragma unroll
  for (int r = 0; r < 2; ++r) {
#pragma unroll
    for (int j = 0; j < 4; ++j) {
      const int col = colbase + j * 16 + c;
      const float bv = bias[col];
#pragma unroll
      for (int e = 0; e < 8; ++e) {
        const int row   = rowbase + r * 16 + e + 8 * hi;   // D layout
        const float val = (acc[r][j][e] + bv) * scale;
        if (OUT_MODE == 0) {
          ((_Float16*)outv)[(size_t)row * DIM + col] = (_Float16)val;
        } else if (OUT_MODE == 1) {
          const int b = row >> 10, m = row & 1023;
          ((_Float16*)outv)[((size_t)b * DIM + col) * 1024 + m] = (_Float16)val;
        } else {
          ((float*)outv)[(size_t)row * DIM + col] = val;
        }
      }
    }
  }
}

// ---------------------------------------------------------------------------
// Flash attention. Block = 4 waves; each wave owns 16 q-rows of one (b,h);
// kv streamed in steps of 64. K tile [key][d] and V tile [d][key] (8KB each)
// are async-copied into double-buffered LDS shared by the 4 waves. Online
// softmax in exp2 domain (log2e folded into Q); P goes D-layout -> A-layout
// through a private per-wave LDS tile.
// ---------------------------------------------------------------------------
__global__ __launch_bounds__(128) void attn_kernel(
    const _Float16* __restrict__ q, const _Float16* __restrict__ k,
    const _Float16* __restrict__ vT, _Float16* __restrict__ attn) {
  __shared__ __align__(16) _Float16 kbuf[2][64][64];   // [buf][key][d]
  __shared__ __align__(16) _Float16 vbuf[2][64][64];   // [buf][d][key]
  __shared__ __align__(16) _Float16 pbuf[4][16][64];   // per-wave P staging

  const int wave = threadIdx.x >> 5;
  const int lane = threadIdx.x & 31;
  const int b    = blockIdx.z;
  const int h    = blockIdx.y;
  const int qrow = blockIdx.x * 64 + wave * 16;        // row within batch
  const int hi   = lane >> 4;

  // async-staging assignment: 128 threads x (4+4) b128 per kv-step
  const int srow = threadIdx.x >> 1;                   // 0..63
  const int soff = (threadIdx.x & 1) * 32;             // 0 or 32 halfs
  const _Float16* ksrc = k  + ((size_t)(b * 1024 + srow)) * DIM + h * HDIM + soff;
  const _Float16* vsrc = vT + ((size_t)(b * DIM + h * HDIM + srow)) * 1024 + soff;

  const _Float16* qp = q + ((size_t)(b * 1024 + qrow)) * DIM + h * HDIM;
  const v16h q0 = frag_f16(qp, DIM);       // d 0..31 (SCALE*log2e folded in)
  const v16h q1 = frag_f16(qp + 32, DIM);  // d 32..63

  // prologue: stage kv tile 0
#pragma unroll
  for (int i = 0; i < 4; ++i) {
    async_b128(&kbuf[0][srow][soff] + i * 8, ksrc + (size_t)0 * DIM + i * 8);
    async_b128(&vbuf[0][srow][soff] + i * 8, vsrc + 0 + i * 8);
  }
  wait_async0();
  __syncthreads();

  v8f o[4] = {};
  float mrun[8], lrun[8];
#pragma unroll
  for (int e = 0; e < 8; ++e) { mrun[e] = -3.0e38f; lrun[e] = 0.0f; }

  for (int kv = 0; kv < 1024; kv += 64) {
    const int p = (kv >> 6) & 1;
    if (kv + 64 < 1024) {                               // kick next tiles
#pragma unroll
      for (int i = 0; i < 4; ++i) {
        async_b128(&kbuf[p ^ 1][srow][soff] + i * 8,
                   ksrc + (size_t)(kv + 64) * DIM + i * 8);
        async_b128(&vbuf[p ^ 1][srow][soff] + i * 8, vsrc + kv + 64 + i * 8);
      }
    }

    // S = Q(16x64) @ K^T for 4 groups of 16 keys
    v8f s[4];
    const v8f zero = {};
#pragma unroll
    for (int kg = 0; kg < 4; ++kg) {
      v16h bk = frag_f16(&kbuf[p][kg * 16][0], 64);
      s[kg] = wmma_f16(q0, bk, zero);
      bk = frag_f16(&kbuf[p][kg * 16][32], 64);
      s[kg] = wmma_f16(q1, bk, s[kg]);
    }

    // Online softmax (exp2 domain). Row r's 64 scores: element e = r&7 across
    // the 16 lanes of this half-wave; xor masks 1/2/4/8 stay in the group.
#pragma unroll
    for (int e = 0; e < 8; ++e) {
      float mx = fmaxf(fmaxf(s[0][e], s[1][e]), fmaxf(s[2][e], s[3][e]));
      mx = fmaxf(mx, __shfl_xor(mx, 1));
      mx = fmaxf(mx, __shfl_xor(mx, 2));
      mx = fmaxf(mx, __shfl_xor(mx, 4));
      mx = fmaxf(mx, __shfl_xor(mx, 8));
      mx = fmaxf(mx, mrun[e]);
      const float rescale = exp2f(mrun[e] - mx);
      float pv[4];
#pragma unroll
      for (int kg = 0; kg < 4; ++kg) pv[kg] = exp2f(s[kg][e] - mx);
      float rs = (pv[0] + pv[1]) + (pv[2] + pv[3]);
      rs += __shfl_xor(rs, 1);
      rs += __shfl_xor(rs, 2);
      rs += __shfl_xor(rs, 4);
      rs += __shfl_xor(rs, 8);
      lrun[e] = lrun[e] * rescale + rs;
      mrun[e] = mx;
#pragma unroll
      for (int j = 0; j < 4; ++j) o[j][e] *= rescale;
      const int prw = e + 8 * hi;                       // D layout -> P matrix
#pragma unroll
      for (int kg = 0; kg < 4; ++kg)
        pbuf[wave][prw][kg * 16 + (lane & 15)] = (_Float16)pv[kg];
    }

    // O += P(16x64) @ V(64x64)
    const v16h pa0 = frag_f16(&pbuf[wave][0][0], 64);
    const v16h pa1 = frag_f16(&pbuf[wave][0][32], 64);
#pragma unroll
    for (int j = 0; j < 4; ++j) {
      o[j] = wmma_f16(pa0, frag_f16(&vbuf[p][j * 16][0], 64), o[j]);
      o[j] = wmma_f16(pa1, frag_f16(&vbuf[p][j * 16][32], 64), o[j]);
    }

    wait_async0();
    __syncthreads();
  }

  // Normalize and emit in [B*N, H*d] f16 layout for the output projection.
#pragma unroll
  for (int j = 0; j < 4; ++j) {
    const int col = h * HDIM + j * 16 + (lane & 15);
#pragma unroll
    for (int e = 0; e < 8; ++e) {
      const int row = b * 1024 + qrow + e + 8 * hi;
      attn[(size_t)row * DIM + col] = (_Float16)(o[j][e] / lrun[e]);
    }
  }
}

// ---------------------------------------------------------------------------

extern "C" void kernel_launch(void* const* d_in, const int* in_sizes, int n_in,
                              void* d_out, int out_size, void* d_ws, size_t ws_size,
                              hipStream_t stream) {
  const float* x   = (const float*)d_in[0];
  const float* ctx = (const float*)d_in[1];
  const float* Wq  = (const float*)d_in[2];
  const float* bq  = (const float*)d_in[3];
  const float* Wk  = (const float*)d_in[4];
  const float* bk  = (const float*)d_in[5];
  const float* Wv  = (const float*)d_in[6];
  const float* bv  = (const float*)d_in[7];
  const float* Wo  = (const float*)d_in[8];
  const float* bo  = (const float*)d_in[9];

  _Float16* q_ws  = (_Float16*)d_ws;                 // [8192 x 1024] f16 (pre-scaled)
  _Float16* k_ws  = q_ws  + (size_t)NTOK * DIM;      // [8192 x 1024] f16
  _Float16* vT_ws = k_ws  + (size_t)NTOK * DIM;      // per-b transposed f16
  _Float16* a_ws  = vT_ws + (size_t)NTOK * DIM;      // attention out f16
  _Float16* wtq   = a_ws  + (size_t)NTOK * DIM;      // f16 [N x K] weights
  _Float16* wtk   = wtq + (size_t)DIM * DIM;
  _Float16* wtv   = wtk + (size_t)DIM * DIM;
  _Float16* wto   = wtv + (size_t)DIM * DIM;

  const dim3 tg(DIM / 32, DIM / 32), tt(256);
  w_transpose_kernel<<<tg, tt, 0, stream>>>(Wq, wtq);
  w_transpose_kernel<<<tg, tt, 0, stream>>>(Wk, wtk);
  w_transpose_kernel<<<tg, tt, 0, stream>>>(Wv, wtv);
  w_transpose_kernel<<<tg, tt, 0, stream>>>(Wo, wto);

  const dim3 gg(DIM / 64, NTOK / 256), tb(256);
  gemm1024_kernel<false, 0><<<gg, tb, 0, stream>>>(x,   wtq, bq, q_ws,  ATT_SCALE * LOG2E);
  gemm1024_kernel<false, 0><<<gg, tb, 0, stream>>>(ctx, wtk, bk, k_ws,  1.0f);
  gemm1024_kernel<false, 1><<<gg, tb, 0, stream>>>(ctx, wtv, bv, vT_ws, 1.0f);

  attn_kernel<<<dim3(1024 / 64, NHEADS, 8), 128, 0, stream>>>(q_ws, k_ws, vT_ws, a_ws);

  gemm1024_kernel<true, 2><<<gg, tb, 0, stream>>>(a_ws, wto, bo, d_out, 1.0f);
}
